// MoDRouter_48387101556956
// MI455X (gfx1250) — matrix operand order repaired and checked by reading, not compiled, not run on hardware
//
#include <hip/hip_runtime.h>

#define Bsz   4
#define Ntok  4096
#define Dm    1024
#define DFFc  4096
#define KSEL  2048

typedef __attribute__((ext_vector_type(16))) __bf16 bfx16;
typedef __attribute__((ext_vector_type(8)))  float  v8f;

__device__ __forceinline__ v8f wmma_bf16(bfx16 a, bfx16 b, v8f c) {
  // D = A*B + C, fp32 accumulate, emits v_wmma_f32_16x16x32_bf16
  return __builtin_amdgcn_wmma_f32_16x16x32_bf16(false, a, false, b, (short)0, c,
                                                 false, false);
}

__device__ __forceinline__ unsigned short bfbits(float v) {
  return __builtin_bit_cast(unsigned short, (__bf16)v);
}

__device__ __forceinline__ float gelu_tanh(float x) {
  const float c0 = 0.7978845608028654f; // sqrt(2/pi)
  const float c1 = 0.044715f;
  float t = c0 * (x + c1 * x * x * x);
  return 0.5f * x * (1.0f + tanhf(t));
}

// ---------------- out = x ----------------
__global__ void mod_copy(const float4* __restrict__ src, float4* __restrict__ dst,
                         int n) {
  int i = blockIdx.x * blockDim.x + threadIdx.x;
  if (i < n) dst[i] = src[i];
}

// ------------- gate scores: one wave32 per token -------------
__global__ void mod_gate(const float* __restrict__ x, const float* __restrict__ gw,
                         float* __restrict__ scores) {
  int lane = threadIdx.x & 31;
  int wv = threadIdx.x >> 5;
  int t = blockIdx.x * 8 + wv; // global token id in [0, B*N)
  const float* row = x + (size_t)t * Dm;
  float s = 0.f;
  for (int i = lane; i < Dm; i += 32) s += row[i] * gw[i];
#pragma unroll
  for (int off = 16; off > 0; off >>= 1) s += __shfl_xor(s, off, 32);
  if (lane == 0) scores[t] = s;
}

// ------------- per-batch top-k via 4-pass radix select -------------
__device__ __forceinline__ unsigned f2key(float f) {
  unsigned u = __builtin_bit_cast(unsigned, f);
  return (u & 0x80000000u) ? ~u : (u | 0x80000000u);
}

__global__ void mod_topk(const float* __restrict__ scores, int* __restrict__ sel) {
  __shared__ unsigned hist[256];
  __shared__ unsigned sh_prefix;
  __shared__ int sh_remaining;
  __shared__ unsigned sh_cnt;
  const int b = blockIdx.x;
  const int tid = threadIdx.x;
  const float* sc = scores + b * Ntok;

  unsigned prefix = 0;
  int remaining = KSEL;
#pragma unroll 1
  for (int pass = 0; pass < 4; ++pass) {
    int shift = 24 - 8 * pass;
    unsigned maskhi = (pass == 0) ? 0u : (0xFFFFFFFFu << (shift + 8));
    hist[tid] = 0;
    __syncthreads();
    for (int n = tid; n < Ntok; n += 256) {
      unsigned u = f2key(sc[n]);
      if ((u & maskhi) == (prefix & maskhi))
        atomicAdd(&hist[(u >> shift) & 255], 1u);
    }
    __syncthreads();
    if (tid == 0) {
      int rem = remaining;
      int d = 255;
      for (; d > 0; --d) {
        int c = (int)hist[d];
        if (rem <= c) break;
        rem -= c;
      }
      sh_prefix = prefix | ((unsigned)d << shift);
      sh_remaining = rem;
    }
    __syncthreads();
    prefix = sh_prefix;
    remaining = sh_remaining;
    __syncthreads();
  }
  const unsigned T = prefix; // exact k-th largest key
  if (tid == 0) sh_cnt = 0;
  __syncthreads();
  for (int n = tid; n < Ntok; n += 256) {
    if (f2key(sc[n]) > T) {
      unsigned p = atomicAdd(&sh_cnt, 1u);
      sel[b * KSEL + p] = n;
    }
  }
  __syncthreads();
  if (tid == 0) { // ties: smallest indices first (jax tie-break)
    int p = (int)sh_cnt;
    int need = KSEL - p;
    for (int n = 0; n < Ntok && need > 0; ++n) {
      if (f2key(sc[n]) == T) { sel[b * KSEL + p++] = n; --need; }
    }
  }
}

// ------------- weight prep: bf16 hi/lo in WMMA B-fragment layout -------------
// B (32x16 KxN) per lane: n = lane&15, K = (lane>=16?8:0)+(e&7)+(e>=8?16:0), e = 0..15
__global__ void mod_prep_w1(const float* __restrict__ w1, __bf16* __restrict__ wh,
                            __bf16* __restrict__ wl) {
  int idx = blockIdx.x * 256 + threadIdx.x; // [nt(256)][kc(32)][lane(32)][e(16)]
  int e = idx & 15;
  int lane = (idx >> 4) & 31;
  int kc = (idx >> 9) & 31;
  int nt = idx >> 14;
  int Kl = ((lane >= 16) ? 8 : 0) + (e & 7) + ((e >= 8) ? 16 : 0);
  int d = kc * 32 + Kl;
  int f = nt * 16 + (lane & 15);
  float v = w1[(size_t)d * DFFc + f];
  __bf16 hi = (__bf16)v;
  wh[idx] = hi;
  wl[idx] = (__bf16)(v - (float)hi);
}

__global__ void mod_prep_w2(const float* __restrict__ w2, __bf16* __restrict__ wh,
                            __bf16* __restrict__ wl) {
  int idx = blockIdx.x * 256 + threadIdx.x; // [nt(64)][kc(128)][lane(32)][e(16)]
  int e = idx & 15;
  int lane = (idx >> 4) & 31;
  int kc = (idx >> 9) & 127;
  int nt = idx >> 16;
  int Kl = ((lane >= 16) ? 8 : 0) + (e & 7) + ((e >= 8) ? 16 : 0);
  int fr = kc * 32 + Kl;
  int c = nt * 16 + (lane & 15);
  float v = w2[(size_t)fr * Dm + c];
  __bf16 hi = (__bf16)v;
  wh[idx] = hi;
  wl[idx] = (__bf16)(v - (float)hi);
}

// ------------- fused FFN on selected tokens -------------
// 16 tokens per workgroup, 8 waves; wave w owns output cols [w*128, w*128+128)
__global__ void __launch_bounds__(256)
mod_ffn(const float* __restrict__ x, const int* __restrict__ sel,
        const __bf16* __restrict__ w1h, const __bf16* __restrict__ w1l,
        const __bf16* __restrict__ w2h, const __bf16* __restrict__ w2l,
        const float* __restrict__ b1, const float* __restrict__ b2,
        float* __restrict__ out) {
  extern __shared__ char smemraw[];
  __bf16* XAh = (__bf16*)smemraw;            // 16384 bf16 (32 KB)
  __bf16* XAl = (__bf16*)(smemraw + 32768);  // 16384 bf16 (32 KB)
  __bf16* HAh = (__bf16*)(smemraw + 65536);  // 2048 bf16 (4 KB)
  __bf16* HAl = (__bf16*)(smemraw + 69632);  // 2048 bf16 (4 KB)
  int* stok = (int*)(smemraw + 73728);       // 16 ids

  const int tid = threadIdx.x;
  const int lane = tid & 31;
  const int wv = tid >> 5;
  const int nloc = lane & 15;
  const int b = blockIdx.x / (KSEL / 16);
  const int mt = blockIdx.x % (KSEL / 16);

  if (tid < 16) stok[tid] = sel[b * KSEL + mt * 16 + tid];
  __syncthreads();

  // gather x rows -> LDS A-fragment layout; write adjacent (even,odd) k as one
  // dword into the hi and lo planes (they share lane2 and consecutive e2)
  for (int i = tid; i < 16 * (Dm / 2); i += 256) {
    int m = i >> 9;              // token row 0..15
    int kp = (i & 511) * 2;      // even k
    const float* row = x + ((size_t)(b * Ntok + stok[m])) * Dm;
    float v0 = row[kp];
    float v1 = row[kp + 1];
    __bf16 h0 = (__bf16)v0, h1 = (__bf16)v1;
    unsigned hp = (unsigned)__builtin_bit_cast(unsigned short, h0) |
                  ((unsigned)__builtin_bit_cast(unsigned short, h1) << 16);
    unsigned lp = (unsigned)bfbits(v0 - (float)h0) |
                  ((unsigned)bfbits(v1 - (float)h1) << 16);
    int kc = kp >> 5, kk = kp & 31;
    int lane2 = m + (((kk >> 3) & 1) << 4);
    int e2 = (kk & 7) + ((kk >= 16) ? 8 : 0); // even
    int base = (kc * 32 + lane2) * 16 + e2;
    ((unsigned*)XAh)[base >> 1] = hp;
    ((unsigned*)XAl)[base >> 1] = lp;
  }
  __syncthreads();

  v8f acc[8];
#pragma unroll
  for (int i = 0; i < 8; ++i)
    acc[i] = v8f{0.f, 0.f, 0.f, 0.f, 0.f, 0.f, 0.f, 0.f};

  // Loop-carried per-lane fragment base pointers; all per-tile/per-chunk
  // displacements become constant immediate offsets in the load encodings.
  const __bf16* p1h = w1h + (size_t)wv * 16384 + lane * 16;   // +131072/iter
  const __bf16* p1l = w1l + (size_t)wv * 16384 + lane * 16;
  const __bf16* p2h = w2h + (size_t)wv * 524288 + lane * 16;  // +2048/iter
  const __bf16* p2l = w2l + (size_t)wv * 524288 + lane * 16;
  const __bf16* xah = XAh + lane * 16;  // LDS: kc*512 folds into DS offset
  const __bf16* xal = XAl + lane * 16;
  const __bf16* hah = HAh + lane * 16;
  const __bf16* hal = HAl + lane * 16;

#pragma unroll 1
  for (int iter = 0; iter < DFFc / 128; ++iter) {
    // phase 1: h subtile [16 x 16] at cols iter*128 + wv*16
    v8f h = v8f{0.f, 0.f, 0.f, 0.f, 0.f, 0.f, 0.f, 0.f};
    const float b1v = b1[iter * 128 + wv * 16 + nloc];
#pragma unroll 2
    for (int kc = 0; kc < 32; ++kc) {
      bfx16 ah = *(const bfx16*)(xah + kc * 512);
      bfx16 al = *(const bfx16*)(xal + kc * 512);
      bfx16 bh = *(const bfx16*)(p1h + kc * 512);
      bfx16 bl = *(const bfx16*)(p1l + kc * 512);
      h = wmma_bf16(ah, bh, h);
      h = wmma_bf16(ah, bl, h);
      h = wmma_bf16(al, bh, h);
    }
    __syncthreads(); // prior phase-2 readers of HA are done
#pragma unroll
    for (int r = 0; r < 8; ++r) {
      float g = gelu_tanh(h[r] + b1v);
      int m = r + ((lane >= 16) ? 8 : 0);
      int kl = wv * 16 + nloc;
      int kc4 = kl >> 5, kk = kl & 31;
      int lane2 = m + (((kk >> 3) & 1) << 4);
      int e2 = (kk & 7) + ((kk >= 16) ? 8 : 0);
      int pos = (kc4 * 32 + lane2) * 16 + e2;
      __bf16 gh = (__bf16)g;
      HAh[pos] = gh;
      HAl[pos] = (__bf16)(g - (float)gh);
    }
    __syncthreads();
    // phase 2: hoist the 4 K-chunk A-fragments (hi+lo) once, reuse for 8 tiles
    bfx16 pah[4], pal[4];
#pragma unroll
    for (int kc4 = 0; kc4 < 4; ++kc4) {
      pah[kc4] = *(const bfx16*)(hah + kc4 * 512);
      pal[kc4] = *(const bfx16*)(hal + kc4 * 512);
    }
#pragma unroll
    for (int nt = 0; nt < 8; ++nt) {
#pragma unroll
      for (int kc4 = 0; kc4 < 4; ++kc4) {
        bfx16 bh = *(const bfx16*)(p2h + nt * 65536 + kc4 * 512);
        bfx16 bl = *(const bfx16*)(p2l + nt * 65536 + kc4 * 512);
        acc[nt] = wmma_bf16(pah[kc4], bh, acc[nt]);
        acc[nt] = wmma_bf16(pah[kc4], bl, acc[nt]);
        acc[nt] = wmma_bf16(pal[kc4], bh, acc[nt]);
      }
    }
    p1h += 131072;  // next 8 f-tiles of w1 fragments
    p1l += 131072;
    p2h += 2048;    // next 4 K-chunks of w2 fragments
    p2l += 2048;
  }

  // epilogue: out = x + h2 + b2, scattered to original token rows
#pragma unroll
  for (int nt = 0; nt < 8; ++nt) {
    int col = wv * 128 + nt * 16 + nloc;
    float b2v = b2[col];
#pragma unroll
    for (int r = 0; r < 8; ++r) {
      int m = r + ((lane >= 16) ? 8 : 0);
      size_t off = ((size_t)(b * Ntok + stok[m])) * Dm + col;
      out[off] = x[off] + b2v + acc[nt][r];
    }
  }
}

extern "C" void kernel_launch(void* const* d_in, const int* in_sizes, int n_in,
                              void* d_out, int out_size, void* d_ws, size_t ws_size,
                              hipStream_t stream) {
  (void)in_sizes; (void)n_in; (void)out_size; (void)ws_size;
  const float* x  = (const float*)d_in[0];
  const float* gw = (const float*)d_in[1];
  const float* w1 = (const float*)d_in[2];
  const float* b1 = (const float*)d_in[3];
  const float* w2 = (const float*)d_in[4];
  const float* b2 = (const float*)d_in[5];
  float* out = (float*)d_out;

  char* ws = (char*)d_ws;
  float* scores = (float*)ws;            // 64 KB
  int* sel = (int*)(ws + 65536);         // 32 KB
  __bf16* w1h = (__bf16*)(ws + 131072);  // 4x 8 MB fragment buffers
  __bf16* w1l = w1h + 4194304;
  __bf16* w2h = w1l + 4194304;
  __bf16* w2l = w2h + 4194304;

  const int ncopy = Bsz * Ntok * Dm / 4;
  mod_copy<<<ncopy / 256, 256, 0, stream>>>((const float4*)x, (float4*)out, ncopy);
  mod_prep_w1<<<4194304 / 256, 256, 0, stream>>>(w1, w1h, w1l);
  mod_prep_w2<<<4194304 / 256, 256, 0, stream>>>(w2, w2h, w2l);
  mod_gate<<<(Bsz * Ntok) / 8, 256, 0, stream>>>(x, gw, scores);
  mod_topk<<<Bsz, 256, 0, stream>>>(scores, sel);

  const size_t smem = 73728 + 16 * sizeof(int);
  mod_ffn<<<(Bsz * KSEL) / 16, 256, smem, stream>>>(x, sel, w1h, w1l, w2h, w2l,
                                                    b1, b2, out);
}
